// CascadedPathEncoder_54838142435911
// MI455X (gfx1250) — matrix operand order repaired and checked by reading, not compiled, not run on hardware
//
#include <hip/hip_runtime.h>

// CascadedPathEncoder for MI455X (gfx1250, wave32, WMMA 16x16x32 bf16).
//
// Reference: 32-step scan, each step h = relu([h, wp] @ W[p]^T + b[p]),
// B=8192, H=256, K=260, output = all 32 hidden states -> [8192, 8192] f32.
//
// Design: output write (256 MB) is the roofline floor (~11us @ 23.3 TB/s).
// Compute in bf16 WMMA with f32 accumulate; W (8.5 MB) stays L2-resident;
// recurrent state h lives in LDS per 64-row batch tile so the scan never
// round-trips HBM.

#define NPTS  32
#define PDIM  4
#define HDIM  256
#define KTOT  (HDIM + PDIM)      // 260
#define BATCH 8192
#define MTILE 64                 // batch rows per block
#define HSTR  296                // padded bf16 row stride (K-loop reads up to 287; pad is zero)

typedef __attribute__((ext_vector_type(16))) __bf16 bf16x16;
typedef __attribute__((ext_vector_type(8)))  float  f32x8;

union BReg {
  bf16x16 v;
  uint4   q[2];
  __bf16  h[16];
};

__global__ __launch_bounds__(256)
void cascaded_path_encoder_wmma(const float* __restrict__ path,  // [8192, 128]
                                const float* __restrict__ Wg,    // [32, 256, 260]
                                const float* __restrict__ bg,    // [32, 256]
                                float* __restrict__ out)         // [8192, 8192]
{
  __shared__ __bf16 h_lds[MTILE * HSTR];   // 37,888 bytes

  const int  tid  = threadIdx.x;
  const int  lane = tid & 31;
  const int  wv   = tid >> 5;      // 0..7
  const int  ln16 = lane & 15;
  const bool lo   = lane < 16;
  const int  n0   = wv * 32;       // wave's 32 output columns (2 N-tiles)
  const int  b0   = blockIdx.x * MTILE;

  // Zero hidden state + zero padding (pad stays zero for the whole scan,
  // which makes the 9th K-chunk's A operand exactly zero outside k<260).
  for (int i = tid; i < MTILE * HSTR; i += 256) h_lds[i] = (__bf16)0.0f;
  __syncthreads();

  for (int p = 0; p < NPTS; ++p) {
    // Stage this step's point coords into h_lds[:, 256..259] (concat tail).
    {
      const int m = tid >> 2, c = tid & 3;
      h_lds[m * HSTR + HDIM + c] =
          (__bf16)path[(size_t)(b0 + m) * (NPTS * PDIM) + p * PDIM + c];
    }
    __syncthreads();   // all h/wp writes visible before A-operand reads

    const float* Wp = Wg + (size_t)p * HDIM * KTOT;
    const float* bp = bg + (size_t)p * HDIM;

    // Accumulators seeded with bias: C[m][n] = b[p][n] for every row m.
    f32x8 acc[2][4];
#pragma unroll
    for (int nt = 0; nt < 2; ++nt) {
      const float bias = bp[n0 + nt * 16 + ln16];
#pragma unroll
      for (int ms = 0; ms < 4; ++ms)
#pragma unroll
        for (int j = 0; j < 8; ++j) acc[nt][ms][j] = bias;
    }

    // K loop: 8 full chunks over the hidden state + 1 chunk for the wp tail.
#pragma unroll
    for (int kc = 0; kc < 9; ++kc) {
      const int kb = kc * 32;

      // A operands (16x32 bf16) for the 4 M-subtiles, from LDS.
      // lanes 0-15: elems 0..15 = K kb+0..15
      // lanes 16-31: elems 0..15 = K kb+16..31.
      const int ka = kb + (lo ? 0 : 16);
      BReg A[4];
#pragma unroll
      for (int ms = 0; ms < 4; ++ms) {
        const __bf16* src = &h_lds[(ms * 16 + ln16) * HSTR + ka];
        A[ms].q[0] = *(const uint4*)(src);        // K ka+0..7
        A[ms].q[1] = *(const uint4*)(src + 8);    // K ka+8..15
      }

#pragma unroll
      for (int nt = 0; nt < 2; ++nt) {
        const int n = n0 + nt * 16 + ln16;
        // B operand (32x16 bf16): lane holds 16 consecutive K of one W row.
        // lanes 0-15: K kb+0..15 ; lanes 16-31: K kb+16..31.
        BReg B;
        if (kc < 8) {
          const int k = kb + (lo ? 0 : 16);
          const float4* ws = (const float4*)&Wp[(size_t)n * KTOT + k];
          const float4 f0 = ws[0], f1 = ws[1], f2 = ws[2], f3 = ws[3];
          B.h[0]  = (__bf16)f0.x; B.h[1]  = (__bf16)f0.y;
          B.h[2]  = (__bf16)f0.z; B.h[3]  = (__bf16)f0.w;
          B.h[4]  = (__bf16)f1.x; B.h[5]  = (__bf16)f1.y;
          B.h[6]  = (__bf16)f1.z; B.h[7]  = (__bf16)f1.w;
          B.h[8]  = (__bf16)f2.x; B.h[9]  = (__bf16)f2.y;
          B.h[10] = (__bf16)f2.z; B.h[11] = (__bf16)f2.w;
          B.h[12] = (__bf16)f3.x; B.h[13] = (__bf16)f3.y;
          B.h[14] = (__bf16)f3.z; B.h[15] = (__bf16)f3.w;
        } else {
          // Tail chunk: only K=256..259 are real; avoid OOB past row end.
#pragma unroll
          for (int i = 0; i < 16; ++i) B.h[i] = (__bf16)0.0f;
          if (lo) {
            const float4 f0 = *(const float4*)&Wp[(size_t)n * KTOT + HDIM];
            B.h[0] = (__bf16)f0.x; B.h[1] = (__bf16)f0.y;
            B.h[2] = (__bf16)f0.z; B.h[3] = (__bf16)f0.w;
          }
        }

        // 4 WMMAs reuse this B operand across the M-subtiles.
#pragma unroll
        for (int ms = 0; ms < 4; ++ms)
          acc[nt][ms] = __builtin_amdgcn_wmma_f32_16x16x32_bf16(
              false, A[ms].v, false, B.v, (short)0, acc[nt][ms], false, false);
      }
    }
    __syncthreads();   // all A-operand reads done before h is overwritten

    // Epilogue: ReLU; write f32 to global output and bf16 back to LDS.
#pragma unroll
    for (int nt = 0; nt < 2; ++nt) {
      const int n = n0 + nt * 16 + ln16;
#pragma unroll
      for (int ms = 0; ms < 4; ++ms) {
#pragma unroll
        for (int j = 0; j < 8; ++j) {
          float v = acc[nt][ms][j];
          v = v > 0.0f ? v : 0.0f;
          const int m = ms * 16 + j + (lo ? 0 : 8);
          out[(size_t)(b0 + m) * (NPTS * HDIM) + (size_t)p * HDIM + n] = v;
          h_lds[m * HSTR + n] = (__bf16)v;
        }
      }
    }
    // No barrier needed here: next iteration's wp-store touches a disjoint
    // LDS region, and the __syncthreads() after it fences writes vs reads.
  }
}

extern "C" void kernel_launch(void* const* d_in, const int* in_sizes, int n_in,
                              void* d_out, int out_size, void* d_ws, size_t ws_size,
                              hipStream_t stream) {
  (void)in_sizes; (void)n_in; (void)d_ws; (void)ws_size; (void)out_size;
  const float* path = (const float*)d_in[0];   // [8192, 128]
  const float* W    = (const float*)d_in[1];   // [32, 256, 260]
  const float* b    = (const float*)d_in[2];   // [32, 256]
  float* out        = (float*)d_out;           // [8192, 8192]

  dim3 grid(BATCH / MTILE);   // 128 blocks
  dim3 block(256);            // 8 wave32 waves
  cascaded_path_encoder_wmma<<<grid, block, 0, stream>>>(path, W, b, out);
}
